// HEGN_44160853738071
// MI455X (gfx1250) — matrix operand order repaired and checked by reading, not compiled, not run on hardware
//
#include <hip/hip_runtime.h>
#include <math.h>

typedef __attribute__((ext_vector_type(16))) _Float16 v16h;
typedef __attribute__((ext_vector_type(8)))  float    v8f;

#define EPSF 1e-12f
#define SLOPE 0.2f

// ---------------------------------------------------------------------------
// WMMA GEMM with LDS-staged activations:
//   Y[b,o,s] = sum_c W[o,c] * X[b,c,s]
// W: (O,Cin) row-major.  X: (B,Cin,S).  Y: (B,O,S).
// Requires O == 64, S % 64 == 0.  grid = (S/64, 1, B), block = 256.
// Each block computes a 64x64 output panel.  Per 32-K chunk the block stages
// the 32x64 activation tile into LDS (f16, [col][k] transposed, padded rows):
// each thread converts 8 contiguous k-values and writes one packed
// ds_store_b128; fragment reads are 2x ds_load_b128 per lane.
// ---------------------------------------------------------------------------
__global__ __launch_bounds__(256) void gemm_cmix(
    const float* __restrict__ W, const float* __restrict__ X,
    float* __restrict__ Y, int O, int Cin, int S) {
  __shared__ _Float16 xs[64][40];           // [col][k], padded to 40 (16B-aligned runs)
  int tid  = threadIdx.x;
  int lane = tid & 31;
  int wave = tid >> 5;
  int s4   = blockIdx.x;                    // 64-column panel index
  int b    = blockIdx.z;
  const float* Xb = X + (long)b * Cin * S + (long)s4 * 64;
  float*       Yb = Y + (long)b * O * S + (long)s4 * 64;
  int h  = lane >> 4;
  int ml = lane & 15;
  int ot = wave & 3;                        // output row tile (O==64 -> 4 tiles)
  int cs = (wave >> 2) * 32;                // column half: 0 or 32
  const float* Wrow = W + (long)(ot * 16 + ml) * Cin;

  int ecol = tid & 63;                      // staging: col
  int e8   = (tid >> 6) * 8;                // staging: k base (0,8,16,24)

  v8f acc[2] = {};
  int nchunk = (Cin + 31) >> 5;
  int kend = nchunk * 32;
  for (int kb = 0; kb < kend; kb += 32) {
    bool fullc = (kb + 32 <= Cin);          // block-uniform
    if (fullc) {
      union { _Float16 e[8]; uint4 q; } st;
#pragma unroll
      for (int r = 0; r < 8; ++r)
        st.e[r] = (_Float16)Xb[(long)(kb + e8 + r) * S + ecol];
      *(uint4*)&xs[ecol][e8] = st.q;
      if (kb + 32 < kend)                   // prefetch next chunk (speculative)
        __builtin_prefetch(&Xb[(long)(kb + 32 + e8) * S + ecol], 0, 1);
    } else {
      union { _Float16 e[8]; uint4 q; } st;
#pragma unroll
      for (int r = 0; r < 8; ++r) {
        int c  = kb + e8 + r;
        int cc = c < Cin ? c : Cin - 1;
        float v = Xb[(long)cc * S + ecol];
        st.e[r] = (_Float16)(c < Cin ? v : 0.f);
      }
      *(uint4*)&xs[ecol][e8] = st.q;
    }
    __syncthreads();

    union { _Float16 e[16]; v16h v; } ua;
    if (fullc) {
      const float4* p0 = (const float4*)(Wrow + kb + h * 8);
      const float4* p1 = (const float4*)(Wrow + kb + 16 + h * 8);
      float4 w0 = p0[0], w1 = p0[1], w2 = p1[0], w3 = p1[1];
      ua.e[0]  = (_Float16)w0.x; ua.e[1]  = (_Float16)w0.y;
      ua.e[2]  = (_Float16)w0.z; ua.e[3]  = (_Float16)w0.w;
      ua.e[4]  = (_Float16)w1.x; ua.e[5]  = (_Float16)w1.y;
      ua.e[6]  = (_Float16)w1.z; ua.e[7]  = (_Float16)w1.w;
      ua.e[8]  = (_Float16)w2.x; ua.e[9]  = (_Float16)w2.y;
      ua.e[10] = (_Float16)w2.z; ua.e[11] = (_Float16)w2.w;
      ua.e[12] = (_Float16)w3.x; ua.e[13] = (_Float16)w3.y;
      ua.e[14] = (_Float16)w3.z; ua.e[15] = (_Float16)w3.w;
    } else {
#pragma unroll
      for (int j = 0; j < 16; ++j) {
        int kk = (j < 8) ? (j + h * 8) : (j + 8 + h * 8);
        int c  = kb + kk;
        int cc = c < Cin ? c : Cin - 1;
        float v = Wrow[cc];
        ua.e[j] = (_Float16)(c < Cin ? v : 0.f);
      }
    }
#pragma unroll
    for (int t = 0; t < 2; ++t) {
      union { _Float16 e[16]; v16h v; } ub;
      int col = cs + t * 16 + ml;
#pragma unroll
      for (int j = 0; j < 16; ++j) ub.e[j] = xs[col][h * 8 + j + (j < 8 ? 0 : 8)];
      acc[t] = __builtin_amdgcn_wmma_f32_16x16x32_f16(false, ua.v, false, ub.v,
                                                      (short)0, acc[t], false, false);
    }
    __syncthreads();
  }
#pragma unroll
  for (int t = 0; t < 2; ++t) {
    union { float f[8]; v8f v; } uc; uc.v = acc[t];
#pragma unroll
    for (int i = 0; i < 8; ++i)
      Yb[(long)(ot * 16 + h * 8 + i) * S + cs + t * 16 + ml] = uc.f[i];
  }
}

// ---------------------------------------------------------------------------
// WMMA Gram / negative distance with LDS-staged A strip:
//   G[b,n,m] = 2*<x_n,x_m> - xx[n] - xx[m]
// XF: (B,D,N).  XX: (B,N).  G: (B,N,N).
// Requires N % 512 == 0.  grid = ((N/16)*(N/512), 1, B), block = 256.
// Block computes 16 rows x 512 cols; the shared 16x32 A strip is staged in
// LDS (packed ds_store_b32 per thread) once per chunk and re-read by all
// 8 waves (4 accumulators each).
// ---------------------------------------------------------------------------
__global__ __launch_bounds__(256) void gram_neg(
    const float* __restrict__ XF, const float* __restrict__ XX,
    float* __restrict__ G, int D, int N) {
  __shared__ _Float16 as[16][40];           // [m][k], padded
  int tid  = threadIdx.x;
  int lane = tid & 31;
  int wave = tid >> 5;
  int ncg  = N >> 9;                        // 512-col groups
  int nt   = blockIdx.x / ncg;
  int cg   = blockIdx.x % ncg;
  int b    = blockIdx.z;
  const float* Xb = XF + (long)b * D * N;
  int h  = lane >> 4;
  int ml = lane & 15;
  int colw = cg * 512 + wave * 64;          // this wave's 64-col base
  const float* Bm = Xb + colw + ml;         // + t*16 per column tile

  int em = tid & 15;                        // staging: row within A strip
  int e2 = (tid >> 4) * 2;                  // staging: k base (0..30, step 2)

  v8f acc[4] = {};
  int nchunk = (D + 31) >> 5;
  for (int kb = 0; kb < nchunk * 32; kb += 32) {
    bool fullc = (kb + 32 <= D);            // block-uniform
    if (fullc) {
      union { _Float16 e[2]; unsigned u; } st;
      st.e[0] = (_Float16)Xb[(long)(kb + e2) * N + nt * 16 + em];
      st.e[1] = (_Float16)Xb[(long)(kb + e2 + 1) * N + nt * 16 + em];
      *(unsigned*)&as[em][e2] = st.u;
    } else {
      union { _Float16 e[2]; unsigned u; } st;
      int c0 = kb + e2, c1 = kb + e2 + 1;
      int cc0 = c0 < D ? c0 : D - 1;
      int cc1 = c1 < D ? c1 : D - 1;
      float v0 = Xb[(long)cc0 * N + nt * 16 + em];
      float v1 = Xb[(long)cc1 * N + nt * 16 + em];
      st.e[0] = (_Float16)(c0 < D ? v0 : 0.f);
      st.e[1] = (_Float16)(c1 < D ? v1 : 0.f);
      *(unsigned*)&as[em][e2] = st.u;
    }
    __syncthreads();

    union { _Float16 e[16]; v16h v; } ua;
#pragma unroll
    for (int j = 0; j < 16; ++j) {
      int kk = h * 16 + j;
      ua.e[j] = as[ml][kk];
    }
    if (fullc) {
#pragma unroll
      for (int t = 0; t < 4; ++t) {
        union { _Float16 e[16]; v16h v; } ub;
        const float* xc = Bm + t * 16;
#pragma unroll
        for (int j = 0; j < 16; ++j)
          ub.e[j] = (_Float16)xc[(long)(kb + h * 16 + j) * N];
        acc[t] = __builtin_amdgcn_wmma_f32_16x16x32_f16(false, ua.v, false, ub.v,
                                                        (short)0, acc[t], false, false);
      }
    } else {
#pragma unroll
      for (int t = 0; t < 4; ++t) {
        union { _Float16 e[16]; v16h v; } ub;
        const float* xc = Bm + t * 16;
#pragma unroll
        for (int j = 0; j < 16; ++j) {
          int c  = kb + h * 16 + j;
          int cc = c < D ? c : D - 1;
          float v = xc[(long)cc * N];
          ub.e[j] = (_Float16)(c < D ? v : 0.f);
        }
        acc[t] = __builtin_amdgcn_wmma_f32_16x16x32_f16(false, ua.v, false, ub.v,
                                                        (short)0, acc[t], false, false);
      }
    }
    __syncthreads();
  }
  const float* xxb = XX + (long)b * N;
  float* Gb = G + (long)b * N * N;
#pragma unroll
  for (int t = 0; t < 4; ++t) {
    union { float f[8]; v8f v; } uc; uc.v = acc[t];
    int m = colw + t * 16 + ml;
    float xm = xxb[m];
#pragma unroll
    for (int i = 0; i < 8; ++i) {
      int n = nt * 16 + h * 8 + i;
      Gb[(long)n * N + m] = 2.f * uc.f[i] - xxb[n] - xm;
    }
  }
}

// ---------------------------------------------------------------------------
// Elementwise / reduction kernels
// ---------------------------------------------------------------------------
__global__ void sumsq_k(const float* __restrict__ X, float* __restrict__ XX, int D, int N) {
  int n = blockIdx.x * blockDim.x + threadIdx.x;
  int b = blockIdx.y;
  if (n >= N) return;
  const float* Xb = X + (long)b * D * N;
  float s = 0.f;
  for (int d = 0; d < D; ++d) { float v = Xb[(long)d * N + n]; s += v * v; }
  XX[(long)b * N + n] = s;
}

__global__ void topk16_k(const float* __restrict__ G, int* __restrict__ IDX, int N) {
  int n = blockIdx.x * blockDim.x + threadIdx.x;
  int b = blockIdx.y;
  if (n >= N) return;
  const float* row = G + ((long)b * N + n) * N;
  float bv[16]; int bi[16];
  for (int j = 0; j < 16; ++j) { bv[j] = -3.4e38f; bi[j] = 0; }
  for (int m = 0; m < N; ++m) {
    float v = row[m];
    if (v <= bv[15]) continue;
    int j = 15;
    while (j > 0 && bv[j - 1] < v) { bv[j] = bv[j - 1]; bi[j] = bi[j - 1]; --j; }
    bv[j] = v; bi[j] = m;
  }
  for (int j = 0; j < 16; ++j) IDX[((long)b * N + n) * 16 + j] = bi[j];
}

// GF: (B,2Cin,3,N,K) = [nbr-ctr ; ctr]
__global__ void build_gf_k(const float* __restrict__ X, const int* __restrict__ IDX,
                           float* __restrict__ GF, int Cin, int N, int Kk, int total) {
  int i = blockIdx.x * blockDim.x + threadIdx.x;
  if (i >= total) return;
  int k  = i % Kk;
  int n  = (i / Kk) % N;
  int d  = (i / (Kk * N)) % 3;
  int c2 = (i / (Kk * N * 3)) % (2 * Cin);
  int b  = i / (Kk * N * 3 * 2 * Cin);
  int c  = (c2 < Cin) ? c2 : c2 - Cin;
  float ctr = X[((long)(b * Cin + c) * 3 + d) * N + n];
  float v;
  if (c2 < Cin) {
    int m = IDX[((long)b * N + n) * Kk + k];
    v = X[((long)(b * Cin + c) * 3 + d) * N + m] - ctr;
  } else v = ctr;
  GF[i] = v;
}

// VN leaky-relu projection.  X1,Dv,Out: (B,C,3,S)
__global__ void lrelu_combine_k(const float* __restrict__ X1, const float* __restrict__ Dv,
                                float* __restrict__ Out, int C, int S, int total) {
  int i = blockIdx.x * blockDim.x + threadIdx.x;
  if (i >= total) return;                 // total = B*C*S
  int s = i % S;
  int c = (i / S) % C;
  int b = i / (S * C);
  long base = ((long)(b * C + c) * 3) * S + s;
  float x0 = X1[base], x1 = X1[base + S], x2 = X1[base + 2L * S];
  float d0 = Dv[base], d1 = Dv[base + S], d2 = Dv[base + 2L * S];
  float dot = x0 * d0 + x1 * d1 + x2 * d2;
  float dsq = d0 * d0 + d1 * d1 + d2 * d2;
  float f = dot / (dsq + EPSF);
  bool pos = dot >= 0.f;
  float o0 = SLOPE * x0 + (1.f - SLOPE) * (pos ? x0 : (x0 - f * d0));
  float o1 = SLOPE * x1 + (1.f - SLOPE) * (pos ? x1 : (x1 - f * d1));
  float o2 = SLOPE * x2 + (1.f - SLOPE) * (pos ? x2 : (x2 - f * d2));
  Out[base] = o0; Out[base + S] = o1; Out[base + 2L * S] = o2;
}

// chnorm in-place. X: (B,C,3,S)
__global__ void chnorm_ip_k(float* __restrict__ X, int C, int S, int total) {
  int i = blockIdx.x * blockDim.x + threadIdx.x;
  if (i >= total) return;                 // total = B*S
  int s = i % S;
  int b = i / S;
  float cn2 = 0.f;
  for (int c = 0; c < C; ++c) {
    long base = ((long)(b * C + c) * 3) * S + s;
    float a = X[base], bb = X[base + S], cc = X[base + 2L * S];
    cn2 += a * a + bb * bb + cc * cc;    // sum n^2 = sum of squares
  }
  float cn = fmaxf(sqrtf(cn2), EPSF);
  for (int c = 0; c < C; ++c) {
    long base = ((long)(b * C + c) * 3) * S + s;
    float a = X[base], bb = X[base + S], cc = X[base + 2L * S];
    float n = sqrtf(a * a + bb * bb + cc * cc);
    float k = (n / fmaxf(n, EPSF)) / cn;
    X[base] = a * k; X[base + S] = bb * k; X[base + 2L * S] = cc * k;
  }
}

__global__ void mean_lastk_k(const float* __restrict__ X, float* __restrict__ Y,
                             int Kk, int total) {
  int i = blockIdx.x * blockDim.x + threadIdx.x;
  if (i >= total) return;                 // X: (...,K), Y: (...)
  const float* p = X + (long)i * Kk;
  float s = 0.f;
  for (int k = 0; k < Kk; ++k) s += p[k];
  Y[i] = s / (float)Kk;
}

// qk[b,c,n,k] = sum_d Ky * Qx
__global__ void qk_k(const float* __restrict__ KY, const float* __restrict__ QX,
                     float* __restrict__ QK, int N, int Kk, int total) {
  int i = blockIdx.x * blockDim.x + threadIdx.x;
  if (i >= total) return;
  int k = i % Kk;
  int n = (i / Kk) % N;
  int c = (i / (Kk * N)) % 64;
  int b = i / (Kk * N * 64);
  float s = 0.f;
  for (int d = 0; d < 3; ++d)
    s += KY[(((long)(b * 64 + c) * 3 + d) * N + n) * Kk + k] *
         QX[((long)(b * 64 + c) * 3 + d) * N + n];
  QK[i] = s;
}

__global__ void softmax_heads_k(const float* __restrict__ QK, float* __restrict__ ATT,
                                int N, int Kk, int total) {
  int i = blockIdx.x * blockDim.x + threadIdx.x;
  if (i >= total) return;                 // total = B*2*N
  int n  = i % N;
  int hh = (i / N) % 2;
  int b  = i / (N * 2);
  float lg[16];
  float mx = -3.4e38f;
  for (int k = 0; k < Kk; ++k) {
    float s = 0.f;
    for (int c = 0; c < 32; ++c) {
      int cc = hh * 32 + c;
      s += QK[((long)(b * 64 + cc) * N + n) * Kk + k];
    }
    s *= 0.1020620726f;                   // 1/sqrt(3*MHC)
    lg[k] = s; mx = fmaxf(mx, s);
  }
  float den = 0.f;
  for (int k = 0; k < Kk; ++k) { lg[k] = expf(lg[k] - mx); den += lg[k]; }
  for (int k = 0; k < Kk; ++k) ATT[((long)(b * 2 + hh) * N + n) * Kk + k] = lg[k] / den;
}

__global__ void attn_out_k(const float* __restrict__ X, const float* __restrict__ ATT,
                           const float* __restrict__ VY, float* __restrict__ Y,
                           int N, int Kk, int total) {
  int i = blockIdx.x * blockDim.x + threadIdx.x;
  if (i >= total) return;                 // (B,64,3,N)
  int n = i % N;
  int d = (i / N) % 3;
  int c = (i / (N * 3)) % 64;
  int b = i / (N * 3 * 64);
  float s = X[i];
  const float* a = ATT + ((long)(b * 2 + (c >> 5)) * N + n) * Kk;
  const float* v = VY + (((long)(b * 64 + c) * 3 + d) * N + n) * Kk;
  for (int k = 0; k < Kk; ++k) s += a[k] * v[k];
  Y[i] = s;
}

__global__ void mean_lastn_k(const float* __restrict__ X, float* __restrict__ Y,
                             int N, int total) {
  int i = blockIdx.x * blockDim.x + threadIdx.x;
  if (i >= total) return;
  const float* p = X + (long)i * N;
  float s = 0.f;
  for (int n = 0; n < N; ++n) s += p[n];
  Y[i] = s / (float)N;
}

// CAT: (B,128,3,N) = [X ; broadcast(M)]
__global__ void catk_k(const float* __restrict__ X, const float* __restrict__ M,
                       float* __restrict__ CAT, int N, int total) {
  int i = blockIdx.x * blockDim.x + threadIdx.x;
  if (i >= total) return;
  int n  = i % N;
  int d  = (i / N) % 3;
  int c2 = (i / (N * 3)) % 128;
  int b  = i / (N * 3 * 128);
  float v = (c2 < 64) ? X[((long)(b * 64 + c2) * 3 + d) * N + n]
                      : M[(b * 64 + (c2 - 64)) * 3 + d];
  CAT[i] = v;
}

__global__ void meanC_k(const float* __restrict__ X, float* __restrict__ P,
                        int N, int total) {
  int i = blockIdx.x * blockDim.x + threadIdx.x;
  if (i >= total) return;                 // total = B*3*N
  int n = i % N;
  int d = (i / N) % 3;
  int b = i / (N * 3);
  float s = 0.f;
  for (int c = 0; c < 64; ++c) s += X[((long)(b * 64 + c) * 3 + d) * N + n];
  P[i] = s / 64.f;                        // (B,3,N)
}

__global__ void parnorm_k(const float* __restrict__ P, float* __restrict__ PN,
                          int N3, int B, int off) {
  int b = threadIdx.x;
  if (b >= B) return;
  float s = 0.f;
  for (int i = 0; i < N3; ++i) { float v = P[(long)b * N3 + i]; s += v * v; }
  PN[off + b] = sqrtf(s);
}

__global__ void sc_logit_k(const float* __restrict__ FX, const float* __restrict__ FY,
                           const float* __restrict__ PX, const float* __restrict__ PY,
                           const float* __restrict__ PN, float* __restrict__ OUT,
                           int N, int B, int total) {
  int i = blockIdx.x * blockDim.x + threadIdx.x;
  if (i >= total) return;                 // total = B*N
  int n = i % N;
  int b = i / N;
  float nx = PN[b], ny = PN[B + b];
  float acc = 0.f;
  for (int c = 0; c < 64; ++c) {
    float px = 0.f, py = 0.f;
    for (int d = 0; d < 3; ++d) {
      px += FX[((long)(b * 64 + c) * 3 + d) * N + n] * PX[((long)b * 3 + d) * N + n];
      py += FY[((long)(b * 64 + c) * 3 + d) * N + n] * PY[((long)b * 3 + d) * N + n];
    }
    acc += (px / nx) * (py / ny);
  }
  OUT[i] = acc;
}

__global__ void rank_select_k(const float* __restrict__ L, int* __restrict__ SEL,
                              int N, int half) {
  int n = blockIdx.x * blockDim.x + threadIdx.x;
  int b = blockIdx.y;
  if (n >= N) return;
  float v = L[(long)b * N + n];
  int r = 0;
  for (int m = 0; m < N; ++m) {
    float u = L[(long)b * N + m];
    if (u > v || (u == v && m < n)) ++r;
  }
  if (r < half) SEL[(long)b * half + r] = n;
}

__global__ void gatherN_k(const float* __restrict__ X, const int* __restrict__ SEL,
                          float* __restrict__ Y, int N, int half, int total) {
  int i = blockIdx.x * blockDim.x + threadIdx.x;
  if (i >= total) return;                 // (B,64,3,half)
  int j = i % half;
  int d = (i / half) % 3;
  int c = (i / (half * 3)) % 64;
  int b = i / (half * 3 * 64);
  Y[i] = X[((long)(b * 64 + c) * 3 + d) * N + SEL[(long)b * half + j]];
}

__global__ void blockmean_k(const float* __restrict__ X, float* __restrict__ BLK,
                            int Ns, int blk, int total) {
  int i = blockIdx.x * blockDim.x + threadIdx.x;
  if (i >= total) return;                 // total = B*64*3
  int d = i % 3;
  int c = (i / 3) % 64;
  int b = i / (3 * 64);
  const float* p = X + ((long)(b * 64 + c) * 3 + d) * Ns;
  float s = 0.f;
  for (int n = 0; n < Ns; ++n) s += p[n];
  BLK[((long)(b * 128 + blk * 64 + c)) * 3 + d] = s / (float)Ns;
}

__global__ void smallgemm_k(const float* __restrict__ W, const float* __restrict__ X,
                            float* __restrict__ Y, int O, int Cin, int total) {
  int i = blockIdx.x * blockDim.x + threadIdx.x;
  if (i >= total) return;                 // total = B*O*3
  int d = i % 3;
  int o = (i / 3) % O;
  int b = i / (3 * O);
  float s = 0.f;
  for (int c = 0; c < Cin; ++c) s += W[o * Cin + c] * X[((long)(b * Cin + c)) * 3 + d];
  Y[((long)(b * O + o)) * 3 + d] = s;
}

// 3x3 SVD (one-sided Jacobi) per batch; writes R (B,3,3) then S (B,3)
__global__ void svd_out_k(const float* __restrict__ FX, const float* __restrict__ FY,
                          float* __restrict__ OUT, int B) {
  int b = threadIdx.x;
  if (b >= B) return;
  float A[3][3], V[3][3];
  for (int d = 0; d < 3; ++d)
    for (int e = 0; e < 3; ++e) {
      float s = 0.f;
      for (int c = 0; c < 32; ++c)
        s += FX[((long)(b * 32 + c)) * 3 + d] * FY[((long)(b * 32 + c)) * 3 + e];
      A[d][e] = s;
      V[d][e] = (d == e) ? 1.f : 0.f;
    }
  for (int sw = 0; sw < 40; ++sw) {
    for (int p = 0; p < 2; ++p)
      for (int q = p + 1; q < 3; ++q) {
        float al = 0.f, be = 0.f, ga = 0.f;
        for (int i = 0; i < 3; ++i) {
          al += A[i][p] * A[i][p];
          be += A[i][q] * A[i][q];
          ga += A[i][p] * A[i][q];
        }
        if (fabsf(ga) < 1e-20f) continue;
        float zeta = (be - al) / (2.f * ga);
        float t = copysignf(1.f, zeta) / (fabsf(zeta) + sqrtf(1.f + zeta * zeta));
        float cth = 1.f / sqrtf(1.f + t * t);
        float sth = cth * t;
        for (int i = 0; i < 3; ++i) {
          float ap = A[i][p], aq = A[i][q];
          A[i][p] = cth * ap - sth * aq;
          A[i][q] = sth * ap + cth * aq;
          float vp = V[i][p], vq = V[i][q];
          V[i][p] = cth * vp - sth * vq;
          V[i][q] = sth * vp + cth * vq;
        }
      }
  }
  float U[3][3];
  for (int j = 0; j < 3; ++j) {
    float s = 0.f;
    for (int i = 0; i < 3; ++i) s += A[i][j] * A[i][j];
    s = sqrtf(s);
    float inv = (s > 1e-20f) ? 1.f / s : 0.f;
    for (int i = 0; i < 3; ++i) U[i][j] = A[i][j] * inv;
  }
  for (int d = 0; d < 3; ++d)
    for (int e = 0; e < 3; ++e) {
      float s = 0.f;
      for (int j = 0; j < 3; ++j) s += U[d][j] * V[e][j];
      OUT[b * 9 + d * 3 + e] = s;
    }
  for (int e = 0; e < 3; ++e) {
    float fx2 = 0.f, fy2 = 0.f;
    for (int c = 0; c < 32; ++c) {
      float a = FX[((long)(b * 32 + c)) * 3 + e];
      float y = FY[((long)(b * 32 + c)) * 3 + e];
      fx2 += a * a; fy2 += y * y;
    }
    OUT[B * 9 + b * 3 + e] = sqrtf(fy2) / sqrtf(fx2);
  }
}

// ---------------------------------------------------------------------------
// Host orchestration
// ---------------------------------------------------------------------------
extern "C" void kernel_launch(void* const* d_in, const int* in_sizes, int n_in,
                              void* d_out, int out_size, void* d_ws, size_t ws_size,
                              hipStream_t stream) {
  (void)in_sizes; (void)n_in; (void)out_size; (void)ws_size;
  const int B = 2, K = 16;
  const float* fx_in = (const float*)d_in[0];       // (B,3,N) == (B,1,3,N)
  const float* fy_in = (const float*)d_in[1];
  const float* Wt[22];
  for (int i = 0; i < 22; ++i) Wt[i] = (const float*)d_in[2 + i];
  // Wt layout per block i (i*10): dg_w,dg_d,q_w,q_d,k_w,k_d,v_w,v_d,g_w,g_d
  // Wt[20]=h_w (32,128), Wt[21]=h_d (32,32)

  float* base = (float*)d_ws;
  size_t off = 0;
  auto alloc = [&](size_t n) {
    float* p = base + off;
    off += (n + 63) & ~(size_t)63;
    return p;
  };
  float* G    = alloc((size_t)B * 2048 * 2048);
  float* XX   = alloc((size_t)B * 2048);
  float* GF   = alloc((size_t)B * 128 * 3 * 2048 * 16);  // also CAT
  float* T1   = alloc((size_t)B * 64 * 3 * 2048 * 16);
  float* T2   = alloc((size_t)B * 64 * 3 * 2048 * 16);
  float* QXb  = alloc((size_t)B * 64 * 3 * 2048);
  float* ATT  = alloc((size_t)B * 2 * 2048 * 16);
  float* FXD  = alloc((size_t)B * 64 * 3 * 2048);
  float* FYD  = alloc((size_t)B * 64 * 3 * 2048);
  float* FXC  = alloc((size_t)B * 64 * 3 * 2048);
  float* FYC  = alloc((size_t)B * 64 * 3 * 2048);
  float* FXG  = alloc((size_t)B * 64 * 3 * 2048);
  float* FYG  = alloc((size_t)B * 64 * 3 * 2048);
  float* FXS  = alloc((size_t)B * 64 * 3 * 1024);
  float* FYS  = alloc((size_t)B * 64 * 3 * 1024);
  float* MX   = alloc((size_t)B * 64 * 3);
  float* MY   = alloc((size_t)B * 64 * 3);
  float* PARX = alloc((size_t)B * 3 * 2048);
  float* PARY = alloc((size_t)B * 3 * 2048);
  float* PN   = alloc(64);
  float* SCL  = alloc((size_t)B * 2048);
  float* BLKX = alloc((size_t)B * 128 * 3);
  float* BLKY = alloc((size_t)B * 128 * 3);
  float* SF1  = alloc((size_t)B * 32 * 3);
  float* SF2  = alloc((size_t)B * 32 * 3);
  float* FXF  = alloc((size_t)B * 32 * 3);
  float* FYF  = alloc((size_t)B * 32 * 3);
  int*   IDX  = (int*)alloc((size_t)B * 2048 * 16);
  int*   SEL  = (int*)alloc((size_t)B * 1024);

  auto cdiv = [](long n) { return (unsigned)((n + 255) / 256); };

  auto launch_gemm = [&](const float* Wm, const float* Xm, float* Ym,
                         int O, int Cin, int S) {
    dim3 g(S / 64, 1, B);                 // one block per 64x64 output panel
    gemm_cmix<<<g, 256, 0, stream>>>(Wm, Xm, Ym, O, Cin, S);
  };

  // vn_lin_lrelu: X (B,Cin,3,Ssp) -> out (B,O,3,Ssp); uses T1,T2 (out may be T1)
  auto vnll = [&](const float* Xm, const float* w, const float* wd,
                  float* outp, int Cin, int O, long Ssp) {
    launch_gemm(w, Xm, T1, O, Cin, (int)(3 * Ssp));
    launch_gemm(wd, T1, T2, O, O, (int)(3 * Ssp));
    long total = (long)B * O * Ssp;
    lrelu_combine_k<<<cdiv(total), 256, 0, stream>>>(T1, T2, outp, O, (int)Ssp, (int)total);
  };

  // KNN graph feature: Y (B,Cin,3,N) -> GF (B,2Cin,3,N,K), via G/IDX
  auto knn_gf = [&](const float* Ym, int Cin, int N) {
    int D = Cin * 3;
    sumsq_k<<<dim3((N + 255) / 256, B), 256, 0, stream>>>(Ym, XX, D, N);
    dim3 g((N / 16) * (N / 512), 1, B);   // 16x512 panels
    gram_neg<<<g, 256, 0, stream>>>(Ym, XX, G, D, N);
    topk16_k<<<dim3((N + 127) / 128, B), 128, 0, stream>>>(G, IDX, N);
    long total = (long)B * 2 * Cin * 3 * N * K;
    build_gf_k<<<cdiv(total), 256, 0, stream>>>(Ym, IDX, GF, Cin, N, K, (int)total);
  };

  auto dgcnn = [&](const float* Xm, int Cin, int N, const float* dgw,
                   const float* dgd, float* outp) {
    knn_gf(Xm, Cin, N);
    vnll(GF, dgw, dgd, T1, 2 * Cin, 64, (long)N * K);
    long total = (long)B * 64 * 3 * N;
    mean_lastk_k<<<cdiv(total), 256, 0, stream>>>(T1, outp, K, (int)total);
  };

  auto cross = [&](const float* Xm, const float* Ym, int N,
                   const float* qw, const float* qd, const float* kw, const float* kd,
                   const float* vw, const float* vd, float* outp) {
    // Qx
    vnll(Xm, qw, qd, QXb, 64, 64, (long)N);
    chnorm_ip_k<<<cdiv((long)B * N), 256, 0, stream>>>(QXb, 64, N, B * N);
    // graph features of Y
    knn_gf(Ym, 64, N);
    // Ky (lands in T1), chnorm in-place
    vnll(GF, kw, kd, T1, 128, 64, (long)N * K);
    chnorm_ip_k<<<cdiv((long)B * N * K), 256, 0, stream>>>(T1, 64, N * K, B * N * K);
    // qk -> T2, softmax -> ATT
    long tqk = (long)B * 64 * N * K;
    qk_k<<<cdiv(tqk), 256, 0, stream>>>(T1, QXb, T2, N, K, (int)tqk);
    long ts = (long)B * 2 * N;
    softmax_heads_k<<<cdiv(ts), 256, 0, stream>>>(T2, ATT, N, K, (int)ts);
    // Vy (lands in T1, overwrites Ky which is consumed)
    vnll(GF, vw, vd, T1, 128, 64, (long)N * K);
    long to = (long)B * 64 * 3 * N;
    attn_out_k<<<cdiv(to), 256, 0, stream>>>(Xm, ATT, T1, outp, N, K, (int)to);
  };

  const float* Xcur = fx_in;
  const float* Ycur = fy_in;
  int Cin = 1, N = 2048;
  for (int blk = 0; blk < 2; ++blk) {
    const float* const* Wb = &Wt[blk * 10];
    dgcnn(Xcur, Cin, N, Wb[0], Wb[1], FXD);
    dgcnn(Ycur, Cin, N, Wb[0], Wb[1], FYD);
    cross(FXD, FYD, N, Wb[2], Wb[3], Wb[4], Wb[5], Wb[6], Wb[7], FXC);
    cross(FYD, FXC, N, Wb[2], Wb[3], Wb[4], Wb[5], Wb[6], Wb[7], FYC);

    // global context: concat with broadcast mean, then vn_lin_lrelu
    long tm = (long)B * 64 * 3;
    mean_lastn_k<<<cdiv(tm), 256, 0, stream>>>(FXC, MX, N, (int)tm);
    long tc = (long)B * 128 * 3 * N;
    catk_k<<<cdiv(tc), 256, 0, stream>>>(FXC, MX, GF, N, (int)tc);
    vnll(GF, Wb[8], Wb[9], FXG, 128, 64, (long)N);
    mean_lastn_k<<<cdiv(tm), 256, 0, stream>>>(FYC, MY, N, (int)tm);
    catk_k<<<cdiv(tc), 256, 0, stream>>>(FYC, MY, GF, N, (int)tc);
    vnll(GF, Wb[8], Wb[9], FYG, 128, 64, (long)N);

    // invariant scores + top-N/2 selection
    long tp = (long)B * 3 * N;
    meanC_k<<<cdiv(tp), 256, 0, stream>>>(FXG, PARX, N, (int)tp);
    meanC_k<<<cdiv(tp), 256, 0, stream>>>(FYG, PARY, N, (int)tp);
    parnorm_k<<<1, 32, 0, stream>>>(PARX, PN, 3 * N, B, 0);
    parnorm_k<<<1, 32, 0, stream>>>(PARY, PN, 3 * N, B, B);
    long tsc = (long)B * N;
    sc_logit_k<<<cdiv(tsc), 256, 0, stream>>>(FXG, FYG, PARX, PARY, PN, SCL, N, B, (int)tsc);
    int half = N / 2;
    rank_select_k<<<dim3((N + 255) / 256, B), 256, 0, stream>>>(SCL, SEL, N, half);
    long tg = (long)B * 64 * 3 * half;
    gatherN_k<<<cdiv(tg), 256, 0, stream>>>(FXG, SEL, FXS, N, half, (int)tg);
    gatherN_k<<<cdiv(tg), 256, 0, stream>>>(FYG, SEL, FYS, N, half, (int)tg);
    blockmean_k<<<cdiv(tm), 256, 0, stream>>>(FXS, BLKX, half, blk, (int)tm);
    blockmean_k<<<cdiv(tm), 256, 0, stream>>>(FYS, BLKY, half, blk, (int)tm);

    Xcur = FXS; Ycur = FYS; Cin = 64; N = half;
  }

  // final small vn_lin_lrelu (32x128, 32x32) + SVD
  long tf = (long)B * 32 * 3;
  smallgemm_k<<<cdiv(tf), 256, 0, stream>>>(Wt[20], BLKX, SF1, 32, 128, (int)tf);
  smallgemm_k<<<cdiv(tf), 256, 0, stream>>>(Wt[21], SF1, SF2, 32, 32, (int)tf);
  lrelu_combine_k<<<cdiv((long)B * 32), 256, 0, stream>>>(SF1, SF2, FXF, 32, 1, B * 32);
  smallgemm_k<<<cdiv(tf), 256, 0, stream>>>(Wt[20], BLKY, SF1, 32, 128, (int)tf);
  smallgemm_k<<<cdiv(tf), 256, 0, stream>>>(Wt[21], SF1, SF2, 32, 32, (int)tf);
  lrelu_combine_k<<<cdiv((long)B * 32), 256, 0, stream>>>(SF1, SF2, FYF, 32, 1, B * 32);
  svd_out_k<<<1, 32, 0, stream>>>(FXF, FYF, (float*)d_out, B);
}
